// ViT_4844723110053
// MI455X (gfx1250) — matrix operand order repaired and checked by reading, not compiled, not run on hardware
//
#include <hip/hip_runtime.h>
#include <math.h>

// ---------------------------------------------------------------- constants
#define BB       16
#define N_TOK    577
#define NPATCH   576
#define DIM      768
#define QKV_DIM  2304
#define HEADS    12
#define DHEAD    64
#define MLP_DIM  3072
#define NCLS     1000
#define PATCH_D  768          // 3*16*16
#define SCALE_F  0.03608439182435161f   // 768^-0.5

typedef __bf16 bf16;
typedef __attribute__((ext_vector_type(16))) __bf16 v16bf;
typedef __attribute__((ext_vector_type(8)))  __bf16 v8bf;
typedef __attribute__((ext_vector_type(8)))  float  v8f;
typedef __attribute__((ext_vector_type(4)))  int    v4i;

#define HAS_ASYNC_LDS __has_builtin(__builtin_amdgcn_global_load_async_to_lds_b128)

__device__ inline v8f wmma_bf16(v16bf a, v16bf b, v8f c) {
  return __builtin_amdgcn_wmma_f32_16x16x32_bf16(false, a, false, b, (short)0, c,
                                                 false, false);
}
__device__ inline v16bf cat8(v8bf lo, v8bf hi) {
  return __builtin_shufflevector(lo, hi, 0,1,2,3,4,5,6,7,8,9,10,11,12,13,14,15);
}

// 16-byte global -> LDS stage; async DMA (ASYNCcnt) when the toolchain has it.
__device__ inline void stage_b128(const bf16* g, bf16* l) {
#if HAS_ASYNC_LDS
  __builtin_amdgcn_global_load_async_to_lds_b128(
      (v4i*)const_cast<bf16*>(g), (v4i*)l, 0, 0);
#else
  *(v8bf*)l = *(const v8bf*)g;
#endif
}
__device__ inline void wait_async_0() {
#if HAS_ASYNC_LDS
 #if __has_builtin(__builtin_amdgcn_s_wait_asynccnt)
  __builtin_amdgcn_s_wait_asynccnt(0);
 #else
  asm volatile("s_wait_asynccnt 0" ::: "memory");
 #endif
#endif
}
__device__ inline void wait_async_1() {
#if HAS_ASYNC_LDS
 #if __has_builtin(__builtin_amdgcn_s_wait_asynccnt)
  __builtin_amdgcn_s_wait_asynccnt(1);
 #else
  asm volatile("s_wait_asynccnt 1" ::: "memory");
 #endif
#endif
}

// ------------------------------------------------ weight convert + transpose
// in:  [L][K][N] fp32 row-major     out: [L][N][K] bf16
__global__ void cvt_transpose(const float* __restrict__ in, bf16* __restrict__ out,
                              int K, int Nn, long total) {
  long idx = (long)blockIdx.x * blockDim.x + threadIdx.x;
  if (idx >= total) return;
  long kn  = (long)K * Nn;
  long mat = idx / kn;
  long rem = idx - mat * kn;
  int  k   = (int)(rem / Nn);
  int  n   = (int)(rem - (long)k * Nn);
  out[mat * kn + (long)n * K + k] = (bf16)in[idx];
}

// ---------------------------------------------------------- patch gathering
__global__ void patch_gather(const float* __restrict__ img, bf16* __restrict__ Xp) {
  long idx = (long)blockIdx.x * blockDim.x + threadIdx.x;
  const long total = (long)BB * NPATCH * PATCH_D;
  if (idx >= total) return;
  int feat = (int)(idx % PATCH_D);
  long rp  = idx / PATCH_D;
  int p = (int)(rp % NPATCH);
  int b = (int)(rp / NPATCH);
  int c  = feat % 3;
  int px = (feat / 3) & 15;
  int py = feat / 48;
  int gy = p / 24, gx = p % 24;
  int y = gy * 16 + py, x = gx * 16 + px;
  Xp[idx] = (bf16)img[(((size_t)b * 3 + c) * 384 + y) * 384 + x];
}

// ------------------------------------------- x = concat(cls, patches) + pos
__global__ void assemble_x(const float* __restrict__ xt, const float* __restrict__ cls,
                           const float* __restrict__ pos, float* __restrict__ x) {
  long idx = (long)blockIdx.x * blockDim.x + threadIdx.x;
  const long total = (long)BB * N_TOK * DIM;
  if (idx >= total) return;
  int d  = (int)(idx % DIM);
  long bn = idx / DIM;
  int n = (int)(bn % N_TOK);
  int b = (int)(bn / N_TOK);
  float v = (n == 0) ? cls[d] : xt[((size_t)b * NPATCH + (n - 1)) * DIM + d];
  x[idx] = v + pos[(size_t)n * DIM + d];
}

// ------------------------------------------------------------- layer norm
__global__ __launch_bounds__(256)
void ln_kernel(const float* __restrict__ X, size_t inStride,
               const float* __restrict__ w, const float* __restrict__ bvec,
               bf16* __restrict__ out, size_t outStride) {
  __shared__ float red[256];
  const int row = blockIdx.x, tid = threadIdx.x;
  const float* xr = X + (size_t)row * inStride;
  float x0 = xr[tid], x1 = xr[tid + 256], x2 = xr[tid + 512];
  red[tid] = x0 + x1 + x2;
  __syncthreads();
  for (int off = 128; off > 0; off >>= 1) {
    if (tid < off) red[tid] += red[tid + off];
    __syncthreads();
  }
  float mean = red[0] * (1.f / 768.f);
  __syncthreads();
  float d0 = x0 - mean, d1 = x1 - mean, d2 = x2 - mean;
  red[tid] = d0 * d0 + d1 * d1 + d2 * d2;
  __syncthreads();
  for (int off = 128; off > 0; off >>= 1) {
    if (tid < off) red[tid] += red[tid + off];
    __syncthreads();
  }
  float rstd = rsqrtf(red[0] * (1.f / 768.f) + 1e-5f);
  bf16* orow = out + (size_t)row * outStride;
  orow[tid]       = (bf16)(d0 * rstd * w[tid]       + bvec[tid]);
  orow[tid + 256] = (bf16)(d1 * rstd * w[tid + 256] + bvec[tid + 256]);
  orow[tid + 512] = (bf16)(d2 * rstd * w[tid + 512] + bvec[tid + 512]);
}

#define F_BIAS    1
#define F_GELU    2
#define F_RESID   4
#define F_STOREBF 8

// ------------------------------------------------ cooperative tiled GEMM
// 256 threads = 8 waves; block tile 128(M) x 64(N); wave w owns rows
// m0+w*16..+15.  B tile (32k x 64n) double-buffered in LDS, staged with
// async global->LDS b128 DMA (1 chunk / thread), shared by all 8 waves.
template <int FLAGS>
__global__ __launch_bounds__(256)
void gemm_coop(const bf16* __restrict__ A, const bf16* __restrict__ Bt,
               const float* __restrict__ bias, float* __restrict__ Cf,
               bf16* __restrict__ Cb, int M, int Nn, int K) {
  __shared__ __attribute__((aligned(64))) bf16 Bl[2][64 * 32];
  const int tid  = threadIdx.x;
  const int lane = tid & 31, wave = tid >> 5;
  const int half = lane >> 4, lr = lane & 15;
  const int n0 = blockIdx.x * 64;
  const int m0 = blockIdx.y * 128 + wave * 16;

  // staging: thread -> (col 0..63, 8-elem k-chunk 0..3)
  const int scol = tid >> 2, skp = tid & 3;
  int bcol = n0 + scol;
  if (bcol >= Nn) bcol = Nn - 1;                 // clamp loads, guard stores
  const bf16* gB   = Bt + (size_t)bcol * K + skp * 8;
  bf16* lB0 = &Bl[0][scol * 32 + skp * 8];
  bf16* lB1 = &Bl[1][scol * 32 + skp * 8];

  int arow = m0 + lr;
  if (arow >= M) arow = M - 1;
  const bf16* arowp = A + (size_t)arow * K;

  v8f acc[4] = {};
  const int nk = K / 32;

  stage_b128(gB, lB0);                            // prime buffer 0 (k=0)
  for (int i = 0; i < nk; ++i) {
    const int k0 = i * 32;
    const bool hasNext = (i + 1 < nk);
    if (hasNext) stage_b128(gB + k0 + 32, (i & 1) ? lB0 : lB1);
    if (hasNext) wait_async_1(); else wait_async_0();
    __syncthreads();                              // buf (i&1) resident for all waves

    const bf16* ap = arowp + k0 + half * 16;
    v16bf a = *(const v16bf*)ap;
    const bf16* base = Bl[i & 1];
#pragma unroll
    for (int t = 0; t < 4; ++t) {
      const bf16* bp = base + (size_t)(t * 16 + lr) * 32 + half * 16;
      v16bf b = *(const v16bf*)bp;
      acc[t] = wmma_bf16(a, b, acc[t]);
    }
    __syncthreads();                              // done reading buf (i&1)
  }

#pragma unroll
  for (int t = 0; t < 4; ++t) {
    int col = n0 + t * 16 + lr;
    if (col >= Nn) continue;
    float bv = (FLAGS & F_BIAS) ? bias[col] : 0.f;
#pragma unroll
    for (int r = 0; r < 8; ++r) {
      int row = m0 + half * 8 + r;
      if (row >= M) continue;
      float v = acc[t][r] + bv;
      if (FLAGS & F_GELU) v = 0.5f * v * (1.f + erff(v * 0.70710678118654752f));
      size_t idx = (size_t)row * Nn + col;
      if (FLAGS & F_RESID) v += Cf[idx];
      if (FLAGS & F_STOREBF) Cb[idx] = (bf16)v;
      else                   Cf[idx] = v;
    }
  }
}

// ------------------------------------- single-wave GEMM (tiny M, head only)
template <int FLAGS>
__global__ __launch_bounds__(32)
void gemm_wmma(const bf16* __restrict__ A, const bf16* __restrict__ Bt,
               const float* __restrict__ bias, float* __restrict__ Cf,
               bf16* __restrict__ Cb, int M, int Nn, int K) {
  const int lane = threadIdx.x;
  const int half = lane >> 4, lr = lane & 15;
  const int n0 = blockIdx.x * 64;
  const int m0 = blockIdx.y * 16;
  (void)M;

  v8f acc[4] = {};
  int colc[4];
#pragma unroll
  for (int t = 0; t < 4; ++t) {
    int c = n0 + t * 16 + lr;
    colc[t] = c < Nn ? c : Nn - 1;
  }
  const bf16* arow = A + (size_t)(m0 + lr) * K;
  for (int k0 = 0; k0 < K; k0 += 32) {
    const bf16* ap = arow + k0 + half * 16;
    v16bf a = *(const v16bf*)ap;
#pragma unroll
    for (int t = 0; t < 4; ++t) {
      const bf16* bp = Bt + (size_t)colc[t] * K + k0 + half * 16;
      v16bf b = *(const v16bf*)bp;
      acc[t] = wmma_bf16(a, b, acc[t]);
    }
  }
#pragma unroll
  for (int t = 0; t < 4; ++t) {
    int col = n0 + t * 16 + lr;
    if (col >= Nn) continue;
    float bv = (FLAGS & F_BIAS) ? bias[col] : 0.f;
#pragma unroll
    for (int r = 0; r < 8; ++r) {
      int row = m0 + half * 8 + r;
      float v = acc[t][r] + bv;
      if (FLAGS & F_GELU) v = 0.5f * v * (1.f + erff(v * 0.70710678118654752f));
      size_t idx = (size_t)row * Nn + col;
      if (FLAGS & F_RESID) v += Cf[idx];
      if (FLAGS & F_STOREBF) Cb[idx] = (bf16)v;
      else                   Cf[idx] = v;
    }
  }
}

// --------------------------------------------------------- flash attention
// one wave per (b, head, 16-row tile).  qkv: [B*N][2304] bf16.
__global__ __launch_bounds__(32)
void attn_wmma(const bf16* __restrict__ qkv, bf16* __restrict__ O) {
  const int lane = threadIdx.x, half = lane >> 4, lr = lane & 15;
  const int mt = blockIdx.x, h = blockIdx.y, b = blockIdx.z;
  const int m0 = mt * 16;
  __shared__ __attribute__((aligned(32))) bf16 Pl[16][32];

  int qrow = m0 + lr;
  if (qrow > N_TOK - 1) qrow = N_TOK - 1;
  const bf16* qbase = qkv + (size_t)(b * N_TOK + qrow) * QKV_DIM + h * DHEAD;
  const bf16* p0 = qbase + half * 16;
  v16bf aq0 = *(const v16bf*)p0;
  const bf16* p1 = qbase + 32 + half * 16;
  v16bf aq1 = *(const v16bf*)p1;

  v8f o0 = {}, o1 = {}, o2 = {}, o3 = {};
  float mi[8], li[8];
#pragma unroll
  for (int r = 0; r < 8; ++r) { mi[r] = -3.0e38f; li[r] = 0.f; }

  for (int j0 = 0; j0 < N_TOK; j0 += 32) {
    int key0 = j0 + lr, key1 = j0 + 16 + lr;
    int k0c = key0 < N_TOK ? key0 : N_TOK - 1;
    int k1c = key1 < N_TOK ? key1 : N_TOK - 1;
    const bf16* kb0 = qkv + (size_t)(b * N_TOK + k0c) * QKV_DIM + DIM + h * DHEAD + half * 16;
    const bf16* kb1 = qkv + (size_t)(b * N_TOK + k1c) * QKV_DIM + DIM + h * DHEAD + half * 16;
    v16bf b00 = *(const v16bf*)kb0;
    v16bf b01 = *(const v16bf*)(kb0 + 32);
    v16bf b10 = *(const v16bf*)kb1;
    v16bf b11 = *(const v16bf*)(kb1 + 32);

    v8f s0 = {}, s1 = {};
    s0 = wmma_bf16(aq0, b00, s0);
    s0 = wmma_bf16(aq1, b01, s0);
    s1 = wmma_bf16(aq0, b10, s1);
    s1 = wmma_bf16(aq1, b11, s1);

    bool v0 = key0 < N_TOK, v1 = key1 < N_TOK;
    float p0a[8], p1a[8];
#pragma unroll
    for (int r = 0; r < 8; ++r) {
      float x0 = v0 ? s0[r] * SCALE_F : -3.0e38f;
      float x1 = v1 ? s1[r] * SCALE_F : -3.0e38f;
      float m = fmaxf(x0, x1);
      m = fmaxf(m, __shfl_xor(m, 1, 32));
      m = fmaxf(m, __shfl_xor(m, 2, 32));
      m = fmaxf(m, __shfl_xor(m, 4, 32));
      m = fmaxf(m, __shfl_xor(m, 8, 32));
      float mnew  = fmaxf(mi[r], m);
      float alpha = __expf(mi[r] - mnew);
      mi[r] = mnew;
      float e0 = __expf(x0 - mnew), e1 = __expf(x1 - mnew);
      p0a[r] = e0; p1a[r] = e1;
      float rs = e0 + e1;
      rs += __shfl_xor(rs, 1, 32);
      rs += __shfl_xor(rs, 2, 32);
      rs += __shfl_xor(rs, 4, 32);
      rs += __shfl_xor(rs, 8, 32);
      li[r] = li[r] * alpha + rs;
      o0[r] *= alpha; o1[r] *= alpha; o2[r] *= alpha; o3[r] *= alpha;
    }

    __syncthreads();
#pragma unroll
    for (int r = 0; r < 8; ++r) {
      Pl[half * 8 + r][lr]      = (bf16)p0a[r];
      Pl[half * 8 + r][16 + lr] = (bf16)p1a[r];
    }
    __syncthreads();
    v16bf ap = *(const v16bf*)&Pl[lr][half * 16];

    const bf16* vbase = qkv + (size_t)b * N_TOK * QKV_DIM + 2 * DIM + h * DHEAD;
#pragma unroll
    for (int t = 0; t < 4; ++t) {
      v16bf bv;
#pragma unroll
      for (int j = 0; j < 16; ++j) {
        int vr = j0 + half * 16 + j;
        if (vr > N_TOK - 1) vr = N_TOK - 1;
        bv[j] = vbase[(size_t)vr * QKV_DIM + t * 16 + lr];
      }
      v8f& ot = (t == 0 ? o0 : t == 1 ? o1 : t == 2 ? o2 : o3);
      ot = wmma_bf16(ap, bv, ot);
    }
  }

#pragma unroll
  for (int t = 0; t < 4; ++t) {
    v8f ot = (t == 0 ? o0 : t == 1 ? o1 : t == 2 ? o2 : o3);
#pragma unroll
    for (int r = 0; r < 8; ++r) {
      int row = m0 + half * 8 + r;
      if (row < N_TOK)
        O[(size_t)(b * N_TOK + row) * DIM + h * DHEAD + t * 16 + lr] =
            (bf16)(ot[r] / li[r]);
    }
  }
}

// ------------------------------------------------------------------ driver
extern "C" void kernel_launch(void* const* d_in, const int* in_sizes, int n_in,
                              void* d_out, int out_size, void* d_ws, size_t ws_size,
                              hipStream_t stream) {
  (void)in_sizes; (void)n_in; (void)out_size; (void)ws_size;
  const float* img       = (const float*)d_in[0];
  const float* patch_w   = (const float*)d_in[1];
  const float* patch_b   = (const float*)d_in[2];
  const float* pos_emb   = (const float*)d_in[3];
  const float* cls_tok   = (const float*)d_in[4];
  const float* ln1_w     = (const float*)d_in[5];
  const float* ln1_b     = (const float*)d_in[6];
  const float* qkv_w     = (const float*)d_in[7];
  const float* out_w     = (const float*)d_in[8];
  const float* out_b     = (const float*)d_in[9];
  const float* ln2_w     = (const float*)d_in[10];
  const float* ln2_b     = (const float*)d_in[11];
  const float* ff1_w     = (const float*)d_in[12];
  const float* ff1_b     = (const float*)d_in[13];
  const float* ff2_w     = (const float*)d_in[14];
  const float* ff2_b     = (const float*)d_in[15];
  const float* head_ln_w = (const float*)d_in[16];
  const float* head_ln_b = (const float*)d_in[17];
  const float* head_w    = (const float*)d_in[18];
  const float* head_b    = (const float*)d_in[19];
  float* out = (float*)d_out;

  const int M  = BB * N_TOK;    // 9232
  const int Mp = BB * NPATCH;   // 9216

  size_t off = 0;
  auto alloc = [&](size_t bytes) -> void* {
    void* p = (char*)d_ws + off;
    off = (off + bytes + 255) & ~(size_t)255;
    return p;
  };
  bf16* wPatchT = (bf16*)alloc((size_t)DIM * DIM * 2);
  bf16* wQKVT   = (bf16*)alloc((size_t)12 * DIM * QKV_DIM * 2);
  bf16* wOutT   = (bf16*)alloc((size_t)12 * DIM * DIM * 2);
  bf16* wFF1T   = (bf16*)alloc((size_t)12 * DIM * MLP_DIM * 2);
  bf16* wFF2T   = (bf16*)alloc((size_t)12 * MLP_DIM * DIM * 2);
  bf16* wHeadT  = (bf16*)alloc((size_t)DIM * NCLS * 2);
  float* xbuf   = (float*)alloc((size_t)M * DIM * 4);
  bf16* hbuf    = (bf16*)alloc((size_t)M * DIM * 2);       // also patch Xp
  bf16* qkvbuf  = (bf16*)alloc((size_t)M * QKV_DIM * 2);
  bf16* obuf    = (bf16*)alloc((size_t)M * DIM * 2);
  bf16* ffbuf   = (bf16*)alloc((size_t)M * MLP_DIM * 2);   // also fp32 xt
  bf16* clsbuf  = (bf16*)alloc((size_t)BB * DIM * 2);
  float* xt     = (float*)ffbuf;                           // alias (disjoint lifetime)
  bf16* Xp      = hbuf;                                    // alias (disjoint lifetime)

  auto cvt = [&](const float* in, bf16* o, int K, int Nn, int L) {
    long tot = (long)L * K * Nn;
    cvt_transpose<<<dim3((unsigned)((tot + 255) / 256)), 256, 0, stream>>>(in, o, K, Nn, tot);
  };
  cvt(patch_w, wPatchT, DIM, DIM, 1);
  cvt(qkv_w,   wQKVT,   DIM, QKV_DIM, 12);
  cvt(out_w,   wOutT,   DIM, DIM, 12);
  cvt(ff1_w,   wFF1T,   DIM, MLP_DIM, 12);
  cvt(ff2_w,   wFF2T,   MLP_DIM, DIM, 12);
  cvt(head_w,  wHeadT,  DIM, NCLS, 1);

  // patch embedding
  {
    long tot = (long)Mp * PATCH_D;
    patch_gather<<<dim3((unsigned)((tot + 255) / 256)), 256, 0, stream>>>(img, Xp);
    gemm_coop<F_BIAS><<<dim3(DIM / 64, Mp / 128), 256, 0, stream>>>(
        Xp, wPatchT, patch_b, xt, nullptr, Mp, DIM, DIM);
    long tot2 = (long)BB * N_TOK * DIM;
    assemble_x<<<dim3((unsigned)((tot2 + 255) / 256)), 256, 0, stream>>>(
        xt, cls_tok, pos_emb, xbuf);
  }

  const int MG = (M + 127) / 128;   // 73 row-blocks of 128
  for (int l = 0; l < 12; ++l) {
    ln_kernel<<<M, 256, 0, stream>>>(xbuf, DIM, ln1_w + (size_t)l * DIM,
                                     ln1_b + (size_t)l * DIM, hbuf, DIM);
    gemm_coop<F_STOREBF><<<dim3(QKV_DIM / 64, MG), 256, 0, stream>>>(
        hbuf, wQKVT + (size_t)l * DIM * QKV_DIM, nullptr, nullptr, qkvbuf,
        M, QKV_DIM, DIM);
    attn_wmma<<<dim3((N_TOK + 15) / 16, HEADS, BB), 32, 0, stream>>>(qkvbuf, obuf);
    gemm_coop<F_BIAS | F_RESID><<<dim3(DIM / 64, MG), 256, 0, stream>>>(
        obuf, wOutT + (size_t)l * DIM * DIM, out_b + (size_t)l * DIM, xbuf, nullptr,
        M, DIM, DIM);
    ln_kernel<<<M, 256, 0, stream>>>(xbuf, DIM, ln2_w + (size_t)l * DIM,
                                     ln2_b + (size_t)l * DIM, hbuf, DIM);
    gemm_coop<F_BIAS | F_GELU | F_STOREBF><<<dim3(MLP_DIM / 64, MG), 256, 0, stream>>>(
        hbuf, wFF1T + (size_t)l * DIM * MLP_DIM, ff1_b + (size_t)l * MLP_DIM,
        nullptr, ffbuf, M, MLP_DIM, DIM);
    gemm_coop<F_BIAS | F_RESID><<<dim3(DIM / 64, MG), 256, 0, stream>>>(
        ffbuf, wFF2T + (size_t)l * MLP_DIM * DIM, ff2_b + (size_t)l * DIM, xbuf,
        nullptr, M, DIM, MLP_DIM);
  }

  // classifier head on cls rows
  ln_kernel<<<BB, 256, 0, stream>>>(xbuf, (size_t)N_TOK * DIM, head_ln_w, head_ln_b,
                                    clsbuf, DIM);
  gemm_wmma<F_BIAS><<<dim3((NCLS + 63) / 64, BB / 16), 32, 0, stream>>>(
      clsbuf, wHeadT, head_b, out, nullptr, BB, NCLS, DIM);
}
